// Q_DenseLayer_68719477415
// MI455X (gfx1250) — 8-bit matrix operand order repaired and checked by reading, compiled, not run on hardware
//
#include <hip/hip_runtime.h>
#include <hip/hip_bf16.h>


typedef int v8i __attribute__((ext_vector_type(8)));

#define BATCH   32
#define CIN     512
#define HH      32
#define WW      32
#define COUT    128
#define NPIX    (BATCH*HH*WW)
#define XTOTAL  (BATCH*CIN*HH*WW)
#define YTOTAL  (BATCH*COUT*HH*WW)

#define WS_WQ     256
#define WS_BQ     2304
#define WS_SCAL   4352
#define WS_WSC    4608
#define WS_BPACK  8192
#define WS_AQ     602112

__global__ void k_init(unsigned int* hdr) {
  hdr[0] = 0u;
  hdr[1] = 0u;
}

__global__ __launch_bounds__(512) void k_bn_prep(
    const float* __restrict__ gamma, const float* __restrict__ beta,
    const float* __restrict__ mean,  const float* __restrict__ var,
    const float* __restrict__ s_in_p,
    float* __restrict__ wq, float* __restrict__ bq, float* __restrict__ scal) {
  __shared__ float red[512];
  const int t = threadIdx.x;
  const float wbn = gamma[t] / sqrtf(var[t] + 1e-5f);
  red[t] = fabsf(wbn);
  __syncthreads();
  for (int s = 256; s > 0; s >>= 1) {
    if (t < s) red[t] = fmaxf(red[t], red[t + s]);
    __syncthreads();
  }
  const float ws_bn = red[0] * (1.0f / 127.0f);
  const float s_in  = s_in_p[0];
  const float bn_sf = ws_bn * s_in;
  float wi = rintf(wbn / ws_bn);
  wi = fminf(fmaxf(wi, -128.0f), 127.0f);
  const float bbn = beta[t] - mean[t] * wbn;
  wq[t] = wi;
  bq[t] = rintf(bbn / bn_sf);
  if (t == 0) scal[0] = bn_sf;
}

__global__ __launch_bounds__(256) void k_act_max(
    const float* __restrict__ x, const float* __restrict__ s_in_p,
    const float* __restrict__ wq, const float* __restrict__ bq,
    const float* __restrict__ scal, unsigned int* hdr) {
  const float inv_sin = 1.0f / s_in_p[0];
  const float bn_sf = scal[0];
  float lmax = 0.0f;
  for (int i = blockIdx.x * blockDim.x + threadIdx.x; i < XTOTAL;
       i += gridDim.x * blockDim.x) {
    const int c = (i >> 10) & (CIN - 1);
    float a = (x[i] * inv_sin * wq[c] + bq[c]) * bn_sf;
    a = fmaxf(a, 0.0f);
    lmax = fmaxf(lmax, a);
  }
  __shared__ float red[256];
  red[threadIdx.x] = lmax;
  __syncthreads();
  for (int s = 128; s > 0; s >>= 1) {
    if (threadIdx.x < s) red[threadIdx.x] = fmaxf(red[threadIdx.x], red[threadIdx.x + s]);
    __syncthreads();
  }
  if (threadIdx.x == 0) atomicMax(hdr, __float_as_uint(red[0]));
}

__global__ __launch_bounds__(256) void k_act_quant(
    const float* __restrict__ x, const float* __restrict__ s_in_p,
    const float* __restrict__ wq, const float* __restrict__ bq,
    const float* __restrict__ scal, const unsigned int* __restrict__ hdr,
    signed char* __restrict__ aq) {
  __shared__ signed char t8[WW * CIN];
  const int bh = blockIdx.x;
  const int b = bh >> 5, h = bh & 31;
  const float inv_sin = 1.0f / s_in_p[0];
  const float bn_sf = scal[0];
  const float s_act = __uint_as_float(hdr[0]) * (1.0f / 127.0f);
  const float inv_sa = (s_act > 0.0f) ? (1.0f / s_act) : 0.0f;
  for (int l = threadIdx.x; l < CIN * WW; l += 256) {
    const int c = l >> 5, w = l & 31;
    float a = (x[(size_t)(b * CIN + c) * 1024 + h * 32 + w] * inv_sin * wq[c] + bq[c]) * bn_sf;
    a = fmaxf(a, 0.0f);
    float q = rintf(a * inv_sa);
    q = fminf(fmaxf(q, -128.0f), 127.0f);
    t8[w * CIN + c] = (signed char)q;
  }
  __syncthreads();
  uint4* dst = (uint4*)(aq + (size_t)bh * (WW * CIN));
  const uint4* src = (const uint4*)t8;
  for (int o = threadIdx.x; o < (WW * CIN) / 16; o += 256) dst[o] = src[o];
}

__global__ __launch_bounds__(256) void k_wquant(
    const float* __restrict__ w, float* __restrict__ wsc,
    signed char* __restrict__ bpack) {
  const int co = blockIdx.x;
  const float* wc = w + (size_t)co * (CIN * 9);
  float lmax = 0.0f;
  for (int j = threadIdx.x; j < CIN * 9; j += 256) lmax = fmaxf(lmax, fabsf(wc[j]));
  __shared__ float red[256];
  red[threadIdx.x] = lmax;
  __syncthreads();
  for (int s = 128; s > 0; s >>= 1) {
    if (threadIdx.x < s) red[threadIdx.x] = fmaxf(red[threadIdx.x], red[threadIdx.x + s]);
    __syncthreads();
  }
  const float sc = red[0] * (1.0f / 127.0f);
  const float inv = (sc > 0.0f) ? (1.0f / sc) : 0.0f;
  if (threadIdx.x == 0) wsc[co] = sc;
  for (int j = threadIdx.x; j < CIN * 9; j += 256) {
    const int k = j / 9, tap = j - k * 9;
    float q = rintf(wc[j] * inv);
    q = fminf(fmaxf(q, -128.0f), 127.0f);
    bpack[((size_t)(tap * COUT + co) << 9) + k] = (signed char)q;
  }
}

__global__ __launch_bounds__(256) void k_conv_wmma(
    const signed char* __restrict__ aq, const signed char* __restrict__ bpack,
    const float* __restrict__ wsc, const unsigned int* __restrict__ hdr,
    unsigned int* hdr_out, float* __restrict__ y) {
  __shared__ signed char tA[128 * 512];
  const int tid  = threadIdx.x;
  const int lane = tid & 31;
  const int wid  = tid >> 5;
  const int wm   = wid >> 1;
  const int wn   = wid & 1;
  const int half = lane >> 4;
  const int l16  = lane & 15;
  const int m_tile = blockIdx.x * 128;

  v8i acc[2][4];
  {
    v8i z = {};
#pragma unroll
    for (int i = 0; i < 2; ++i)
#pragma unroll
      for (int j = 0; j < 4; ++j) acc[i][j] = z;
  }

  const int ldrow = tid >> 1;
  const int ldoff = (tid & 1) * 256;
  const int p  = m_tile + ldrow;
  const int pb = p >> 10;
  const int ph = (p >> 5) & 31;
  const int pw = p & 31;

  const signed char* arowbase[2];
#pragma unroll
  for (int mf = 0; mf < 2; ++mf)
    arowbase[mf] = &tA[(wm * 32 + mf * 16 + l16) * 512 + half * 16];
  size_t bbase[4];
#pragma unroll
  for (int nf = 0; nf < 4; ++nf)
    bbase[nf] = (((size_t)(wn * 64 + nf * 16 + l16)) << 9) + half * 16;

  for (int tap = 0; tap < 9; ++tap) {
    const int dh = tap / 3 - 1, dw = tap % 3 - 1;
    const int hh = ph + dh, ww = pw + dw;
    const bool valid = ((unsigned)hh < (unsigned)HH) && ((unsigned)ww < (unsigned)WW);
    const signed char* arow =
        aq + ((size_t)(((pb * 32 + (valid ? hh : 0)) * 32 + (valid ? ww : 0))) << 9) + ldoff;

    __syncthreads();
    {
      uint4* dst = (uint4*)(&tA[ldrow * 512 + ldoff]);
      if (valid) {
        const uint4* src = (const uint4*)arow;
#pragma unroll
        for (int i = 0; i < 16; ++i) dst[i] = src[i];
      } else {
        const uint4 z = make_uint4(0, 0, 0, 0);
#pragma unroll
        for (int i = 0; i < 16; ++i) dst[i] = z;
      }
    }
    __syncthreads();

    const signed char* btap = bpack + (((size_t)tap * COUT) << 9);
    for (int ko = 0; ko < 8; ++ko) {
      v8i afrag[2];
#pragma unroll
      for (int mf = 0; mf < 2; ++mf) {
        const signed char* base = arowbase[mf] + ko * 64;
#pragma unroll
        for (int c2 = 0; c2 < 4; ++c2) {
          const uint2 q = *(const uint2*)(base + (c2 >> 1) * 32 + (c2 & 1) * 8);
          afrag[mf][2 * c2 + 0] = (int)q.x;
          afrag[mf][2 * c2 + 1] = (int)q.y;
        }
      }
#pragma unroll
      for (int nf = 0; nf < 4; ++nf) {
        const signed char* bb = btap + bbase[nf] + ko * 64;
        const uint4 g0 = *(const uint4*)(bb);
        const uint4 g1 = *(const uint4*)(bb + 32);
        v8i bfrag;
        bfrag[0] = (int)g0.x; bfrag[1] = (int)g0.y;
        bfrag[2] = (int)g0.z; bfrag[3] = (int)g0.w;
        bfrag[4] = (int)g1.x; bfrag[5] = (int)g1.y;
        bfrag[6] = (int)g1.z; bfrag[7] = (int)g1.w;
#pragma unroll
        for (int mf = 0; mf < 2; ++mf)
          acc[mf][nf] = __builtin_amdgcn_wmma_i32_16x16x64_iu8(
               true, afrag[mf],  true, bfrag,
              acc[mf][nf],  false,  false);
      }
    }
  }

  const float s_act = __uint_as_float(hdr[0]) * (1.0f / 127.0f);
  float lmax = 0.0f;
#pragma unroll
  for (int nf = 0; nf < 4; ++nf) {
    const int n = wn * 64 + nf * 16 + l16;
    const float cs = s_act * wsc[n];
#pragma unroll
    for (int mf = 0; mf < 2; ++mf) {
#pragma unroll
      for (int r = 0; r < 8; ++r) {
        const int m = m_tile + wm * 32 + mf * 16 + half * 8 + r;
        const int b = m >> 10, hw = m & 1023;
        const float v = (float)acc[mf][nf][r] * cs;
        y[(size_t)(b * COUT + n) * 1024 + hw] = v;
        lmax = fmaxf(lmax, fabsf(v));
      }
    }
  }
  atomicMax(hdr_out, __float_as_uint(lmax));
}

__global__ __launch_bounds__(256) void k_out_quant(
    float* __restrict__ y, const unsigned int* __restrict__ hdr) {
  const float s = __uint_as_float(hdr[1]) * (1.0f / 127.0f);
  const float inv = (s > 0.0f) ? (1.0f / s) : 0.0f;
  for (int i = blockIdx.x * blockDim.x + threadIdx.x; i < YTOTAL;
       i += gridDim.x * blockDim.x) {
    float q = rintf(y[i] * inv);
    q = fminf(fmaxf(q, -128.0f), 127.0f);
    y[i] = q * s;
  }
  if (blockIdx.x == 0 && threadIdx.x == 0) y[YTOTAL] = s;
}

extern "C" void kernel_launch(void* const* d_in, const int* in_sizes, int n_in,
                              void* d_out, int out_size, void* d_ws, size_t ws_size,
                              hipStream_t stream) {
  const float* x     = (const float*)d_in[0];
  const float* s_in  = (const float*)d_in[1];
  const float* gamma = (const float*)d_in[2];
  const float* beta  = (const float*)d_in[3];
  const float* mean  = (const float*)d_in[4];
  const float* var   = (const float*)d_in[5];
  const float* cw    = (const float*)d_in[6];

  char* ws = (char*)d_ws;
  unsigned int* hdr   = (unsigned int*)ws;
  float* wq           = (float*)(ws + WS_WQ);
  float* bq           = (float*)(ws + WS_BQ);
  float* scal         = (float*)(ws + WS_SCAL);
  float* wsc          = (float*)(ws + WS_WSC);
  signed char* bpack  = (signed char*)(ws + WS_BPACK);
  signed char* aq     = (signed char*)(ws + WS_AQ);
  float* y            = (float*)d_out;

  k_init<<<1, 1, 0, stream>>>(hdr);
  k_bn_prep<<<1, 512, 0, stream>>>(gamma, beta, mean, var, s_in, wq, bq, scal);
  k_act_max<<<2048, 256, 0, stream>>>(x, s_in, wq, bq, scal, hdr);
  k_act_quant<<<BATCH * HH, 256, 0, stream>>>(x, s_in, wq, bq, scal, hdr, aq);
  k_wquant<<<COUT, 256, 0, stream>>>(cw, wsc, bpack);
  k_conv_wmma<<<NPIX / 128, 256, 0, stream>>>(aq, bpack, wsc, hdr, hdr + 1, y);
  k_out_quant<<<4096, 256, 0, stream>>>(y, hdr);
}
